// TinyViTBlock_32306744000558
// MI455X (gfx1250) — compile-verified, matrix operand order checked by reading, not run on hardware
//
#include <hip/hip_runtime.h>
#include <hip/hip_bf16.h>

typedef unsigned short u16;
typedef unsigned int   u32;

typedef __attribute__((ext_vector_type(16))) __bf16 bf16x16;
typedef __attribute__((ext_vector_type(8)))  float  f32x8;

union FragU { bf16x16 v; uint4 q[2]; u32 u[8]; };

// ---------- constants ----------
#define NTOK   50176      // 64 * 784 tokens (also 1024 windows * 49)
#define CDIM   384
#define NHEAD  12
#define KDH    32
#define HIDD   1536
#define WINN   49
#define NWIN   1024
#define SCALE_QK 0.17677669529663687f   // 32^-0.5

// workspace byte offsets
#define OFF_QKV 0
#define SZ_QKV  ((size_t)NTOK * 1152 * 2)
#define OFF_XN  (SZ_QKV)                         // also o (overlay), also tail of h
#define SZ_XN   ((size_t)NTOK * CDIM * 2)
#define OFF_X1  (OFF_XN + SZ_XN)
#define SZ_X1   ((size_t)NTOK * CDIM * 4)
#define OFF_X2  (OFF_X1 + SZ_X1)
#define OFF_M   (OFF_X2 + SZ_X1)

__device__ __forceinline__ u16 f2bf(float f) {
    u32 u = __float_as_uint(f);
    u32 r = u + 0x7FFFu + ((u >> 16) & 1u);
    return (u16)(r >> 16);
}

// A-layout bf16 fragment (16x32) from LDS. Per lane the 16 bf16 are two
// contiguous 16-byte runs: k in [khalf, khalf+8) and [16+khalf, 16+khalf+8).
// Requires (row*ld + kbase + khalf) % 8 == 0  -> ld, kbase multiples of 8.
__device__ __forceinline__ bf16x16 load_frag(const u16* base, int ld, int row_base,
                                             int lane, int kbase) {
    int r     = row_base + (lane & 15);
    int khalf = (lane < 16) ? 0 : 8;
    const u16* p = base + r * ld + kbase + khalf;
    FragU f;
    f.q[0] = *(const uint4*)(p);        // ds_load_b128
    f.q[1] = *(const uint4*)(p + 16);   // ds_load_b128
    return f.v;
}

__device__ __forceinline__ f32x8 wmma_bf16(bf16x16 a, bf16x16 b, f32x8 c) {
    return __builtin_amdgcn_wmma_f32_16x16x32_bf16(false, a, false, b, (short)0, c,
                                                   false, false);
}

// ---------------- LayerNorm (optionally window-permuted output, fp32 -> bf16) --------
__global__ __launch_bounds__(128) void ln_kernel(const float* __restrict__ x,
                                                 const float* __restrict__ g,
                                                 const float* __restrict__ bta,
                                                 u16* __restrict__ out, int permute) {
    int t = blockIdx.x, tid = threadIdx.x;
    const float* xr = x + (size_t)t * CDIM;
    float v[3], s = 0.f;
#pragma unroll
    for (int i = 0; i < 3; ++i) { v[i] = xr[tid * 3 + i]; s += v[i]; }
    __shared__ float red[128];
    red[tid] = s; __syncthreads();
    for (int o = 64; o > 0; o >>= 1) { if (tid < o) red[tid] += red[tid + o]; __syncthreads(); }
    float mean = red[0] * (1.f / CDIM);
    __syncthreads();
    float s2 = 0.f;
#pragma unroll
    for (int i = 0; i < 3; ++i) { float d = v[i] - mean; s2 += d * d; }
    red[tid] = s2; __syncthreads();
    for (int o = 64; o > 0; o >>= 1) { if (tid < o) red[tid] += red[tid + o]; __syncthreads(); }
    float inv = rsqrtf(red[0] * (1.f / CDIM) + 1e-5f);

    size_t row;
    if (permute) {
        int b = t / 784, l = t % 784;
        int hh = l / 28, ww = l % 28;
        int wy = hh / 7, py = hh % 7, wx = ww / 7, px = ww % 7;
        row = (size_t)(((b * 4 + wy) * 4 + wx) * WINN + py * 7 + px);
    } else row = (size_t)t;
#pragma unroll
    for (int i = 0; i < 3; ++i) {
        int c = tid * 3 + i;
        out[row * CDIM + c] = f2bf((v[i] - mean) * inv * g[c] + bta[c]);
    }
}

// ---------------- bf16 WMMA GEMM, 128x128x32 block, 8 waves, double-buffered -------
// MODE 0: out bf16 = acc+bias (QKV)
// MODE 1: out f32  = acc+bias+resid, window->token unpermute (proj)
// MODE 2: out bf16 = gelu(acc+bias) (MLP1)
// MODE 3: out f32  = acc+bias+resid (MLP2 -> final output)
template <int MODE>
__global__ __launch_bounds__(256) void gemm_bf16(const u16* __restrict__ A,
                                                 const float* __restrict__ B,
                                                 const float* __restrict__ bias,
                                                 void* __restrict__ out,
                                                 const float* __restrict__ resid,
                                                 int M, int N, int K) {
    __shared__ u16 sA[2][128 * 40];
    __shared__ u16 sB[2][128 * 40];
    int tid = threadIdx.x, lane = tid & 31, wid = tid >> 5;
    int wm = wid >> 2, wn = wid & 3;      // 2 x 4 wave grid -> each wave 64x32
    int m0 = blockIdx.y * 128, n0 = blockIdx.x * 128;

    f32x8 acc[4][2];
    f32x8 zero = 0.0f;
#pragma unroll
    for (int i = 0; i < 4; ++i)
#pragma unroll
        for (int j = 0; j < 2; ++j) acc[i][j] = zero;

    uint4  ra[2];
    float4 rb[4];

    auto loadA = [&](int k0) {
#pragma unroll
        for (int i = 0; i < 2; ++i) {
            int ch = tid + i * 256;              // 0..511 chunks of 8 bf16
            int r = ch >> 2, cc = (ch & 3) << 3;
            ra[i] = *(const uint4*)(A + (size_t)(m0 + r) * K + k0 + cc);
        }
    };
    auto loadB = [&](int k0) {
#pragma unroll
        for (int i = 0; i < 4; ++i) {
            int ch = tid + i * 256;              // 0..1023 chunks of float4
            int r = ch >> 5, cc = (ch & 31) << 2;
            rb[i] = *(const float4*)(B + (size_t)(k0 + r) * N + n0 + cc);
        }
    };
    auto storeTiles = [&](int buf) {
#pragma unroll
        for (int i = 0; i < 2; ++i) {
            int ch = tid + i * 256;
            int r = ch >> 2, cc = (ch & 3) << 3;
            *(uint4*)(&sA[buf][r * 40 + cc]) = ra[i];
        }
#pragma unroll
        for (int i = 0; i < 4; ++i) {
            int ch = tid + i * 256;
            int r = ch >> 5, cc = (ch & 31) << 2;
            sB[buf][(cc + 0) * 40 + r] = f2bf(rb[i].x);
            sB[buf][(cc + 1) * 40 + r] = f2bf(rb[i].y);
            sB[buf][(cc + 2) * 40 + r] = f2bf(rb[i].z);
            sB[buf][(cc + 3) * 40 + r] = f2bf(rb[i].w);
        }
    };

    int nk = K >> 5;
    loadA(0); loadB(0);
    storeTiles(0);

    for (int kt = 0; kt < nk; ++kt) {
        __syncthreads();
        int cur = kt & 1;
        if (kt + 1 < nk) {                       // issue next tile's global loads
            int k0 = (kt + 1) << 5;
            loadA(k0); loadB(k0);
            __builtin_prefetch(A + (size_t)(m0 + (tid >> 1)) * K + k0 + 32, 0, 0);
        }
        // compute current tile: 6 ds_load_b128 pairs + 8 wmma
        bf16x16 af[4], bfg[2];
#pragma unroll
        for (int i = 0; i < 4; ++i) af[i] = load_frag(sA[cur], 40, wm * 64 + i * 16, lane, 0);
#pragma unroll
        for (int j = 0; j < 2; ++j) bfg[j] = load_frag(sB[cur], 40, wn * 32 + j * 16, lane, 0);
#pragma unroll
        for (int i = 0; i < 4; ++i)
#pragma unroll
            for (int j = 0; j < 2; ++j) acc[i][j] = wmma_bf16(af[i], bfg[j], acc[i][j]);
        if (kt + 1 < nk) storeTiles(cur ^ 1);    // park next tile behind the WMMAs
    }

    // epilogue (C/D layout: vgpr r -> M = r or r+8, lane%16 -> N, lane>=16 selects M+8)
#pragma unroll
    for (int i = 0; i < 4; ++i)
#pragma unroll
        for (int j = 0; j < 2; ++j)
#pragma unroll
            for (int r = 0; r < 8; ++r) {
                int ml = wm * 64 + i * 16 + ((lane < 16) ? r : r + 8);
                int nl = wn * 32 + j * 16 + (lane & 15);
                int gm = m0 + ml, gn = n0 + nl;
                float v = acc[i][j][r] + bias[gn];
                if constexpr (MODE == 0) {
                    ((u16*)out)[(size_t)gm * N + gn] = f2bf(v);
                } else if constexpr (MODE == 1) {
                    int w = gm / WINN, p = gm % WINN;
                    int b = w >> 4, wy = (w >> 2) & 3, wx = w & 3;
                    int py = p / 7, px = p % 7;
                    int t = b * 784 + (wy * 7 + py) * 28 + (wx * 7 + px);
                    ((float*)out)[(size_t)t * CDIM + gn] =
                        v + resid[(size_t)t * CDIM + gn];
                } else if constexpr (MODE == 2) {
                    float ge = 0.5f * v * (1.0f + erff(v * 0.70710678118f));
                    ((u16*)out)[(size_t)gm * N + gn] = f2bf(ge);
                } else {
                    ((float*)out)[(size_t)gm * N + gn] = v + resid[(size_t)gm * N + gn];
                }
            }
}

// ---------------- fused window attention: one block per (window, head) ------------
__global__ __launch_bounds__(128) void attn_kernel(const u16* __restrict__ qkv,
                                                   const float* __restrict__ attn_bias,
                                                   const int* __restrict__ bias_idxs,
                                                   u16* __restrict__ o, int n_off) {
    int win = blockIdx.x, hd = blockIdx.y;
    int tid = threadIdx.x, lane = tid & 31, w = tid >> 5;

    __shared__ u16 qs[64 * 40];
    __shared__ u16 ks[64 * 40];
    __shared__ u16 vs[64 * 40];   // V row-major staging
    __shared__ u16 vt[32 * 72];   // V transposed: [d][tok]
    __shared__ u16 pp[64 * 72];   // softmax(P) bf16
    __shared__ float ss[64 * 66]; // scores f32

    const size_t rowbase = (size_t)win * WINN;
    const int cq = hd * 96;       // per-head column base in [1152] (q|k|v = 32|32|32)

    // stage Q/K/V rows with 16B global loads
    for (int idx = tid; idx < 64 * 4; idx += 128) {
        int r = idx >> 2, c0 = (idx & 3) << 3;
        uint4 qv = {0, 0, 0, 0}, kv = {0, 0, 0, 0}, vv = {0, 0, 0, 0};
        if (r < WINN) {
            const u16* g = qkv + (rowbase + r) * 1152 + cq;
            qv = *(const uint4*)(g + c0);
            kv = *(const uint4*)(g + 32 + c0);
            vv = *(const uint4*)(g + 64 + c0);
        }
        *(uint4*)(&qs[r * 40 + c0]) = qv;
        *(uint4*)(&ks[r * 40 + c0]) = kv;
        *(uint4*)(&vs[r * 40 + c0]) = vv;
    }
    __syncthreads();

    // transpose V in LDS (needed only after softmax; overlaps with S phase)
    for (int idx = tid; idx < 32 * 64; idx += 128) {
        int d = idx >> 6, tk = idx & 63;
        vt[d * 72 + tk] = vs[tk * 40 + d];
    }

    // S = Q K^T : wave w owns rows [16w,16w+16)
    {
        bf16x16 aq = load_frag(qs, 40, 16 * w, lane, 0);
        f32x8 accS[4];
        f32x8 zero = 0.0f;
#pragma unroll
        for (int j = 0; j < 4; ++j) {
            bf16x16 bk = load_frag(ks, 40, 16 * j, lane, 0);
            accS[j] = wmma_bf16(aq, bk, zero);
        }
#pragma unroll
        for (int j = 0; j < 4; ++j)
#pragma unroll
            for (int r = 0; r < 8; ++r) {
                int m = 16 * w + ((lane < 16) ? r : r + 8);
                int n = 16 * j + (lane & 15);
                ss[m * 66 + n] = accS[j][r];
            }
    }
    __syncthreads();

    // scaled-bias softmax over valid 49 columns; rows >= 49 zeroed
    if (tid < 64) {
        int nq = tid;
        if (nq < WINN) {
            const int*   ir = bias_idxs + nq * WINN;
            const float* br = attn_bias + (size_t)hd * n_off;
            float mx = -1e30f;
            for (int m2 = 0; m2 < WINN; ++m2) {
                float val = ss[nq * 66 + m2] * SCALE_QK + br[ir[m2]];
                ss[nq * 66 + m2] = val;
                mx = fmaxf(mx, val);
            }
            float sum = 0.f;
            for (int m2 = 0; m2 < WINN; ++m2) {
                float e = __expf(ss[nq * 66 + m2] - mx);
                ss[nq * 66 + m2] = e; sum += e;
            }
            float inv = 1.f / sum;
            for (int m2 = 0; m2 < WINN; ++m2) pp[nq * 72 + m2] = f2bf(ss[nq * 66 + m2] * inv);
            for (int m2 = WINN; m2 < 64; ++m2) pp[nq * 72 + m2] = 0;
        } else {
            for (int m2 = 0; m2 < 64; ++m2) pp[nq * 72 + m2] = 0;
        }
    }
    __syncthreads();

    // O = P V : M tile = wave row strip, N = 32 (2 tiles), K = 64 (2 chunks)
    bf16x16 a0 = load_frag(pp, 72, 16 * w, lane, 0);
    bf16x16 a1 = load_frag(pp, 72, 16 * w, lane, 32);
#pragma unroll
    for (int jn = 0; jn < 2; ++jn) {
        bf16x16 b0 = load_frag(vt, 72, 16 * jn, lane, 0);
        bf16x16 b1 = load_frag(vt, 72, 16 * jn, lane, 32);
        f32x8 accO = 0.0f;
        accO = wmma_bf16(a0, b0, accO);
        accO = wmma_bf16(a1, b1, accO);
#pragma unroll
        for (int r = 0; r < 8; ++r) {
            int m = 16 * w + ((lane < 16) ? r : r + 8);
            if (m < WINN) {
                int n = 16 * jn + (lane & 15);
                o[(rowbase + m) * CDIM + hd * KDH + n] = f2bf(accO[r]);
            }
        }
    }
}

// ---------------- depthwise 3x3 conv + batchnorm (channel-contiguous NHWC) --------
__global__ __launch_bounds__(256) void conv_bn_kernel(const float* __restrict__ x1,
                                                      const float* __restrict__ cw,
                                                      const float* __restrict__ bg,
                                                      const float* __restrict__ bb,
                                                      const float* __restrict__ bm,
                                                      const float* __restrict__ bv,
                                                      float* __restrict__ x2) {
    size_t idx = (size_t)blockIdx.x * 256 + threadIdx.x;
    int c = (int)(idx % CDIM);
    size_t s = idx / CDIM;
    int ww = (int)(s % 28); s /= 28;
    int hh = (int)(s % 28);
    int b  = (int)(s / 28);
    float acc = 0.f;
#pragma unroll
    for (int kh = 0; kh < 3; ++kh) {
        int h2 = hh + kh - 1;
        if (h2 < 0 || h2 > 27) continue;
#pragma unroll
        for (int kw = 0; kw < 3; ++kw) {
            int w2 = ww + kw - 1;
            if (w2 < 0 || w2 > 27) continue;
            acc += x1[(((size_t)b * 28 + h2) * 28 + w2) * CDIM + c] * cw[c * 9 + kh * 3 + kw];
        }
    }
    float sc = bg[c] * rsqrtf(bv[c] + 1e-5f);
    x2[idx] = (acc - bm[c]) * sc + bb[c];
}

// ---------------- launch ----------------
extern "C" void kernel_launch(void* const* d_in, const int* in_sizes, int n_in,
                              void* d_out, int out_size, void* d_ws, size_t ws_size,
                              hipStream_t stream) {
    const float* x        = (const float*)d_in[0];
    const float* ln1_g    = (const float*)d_in[1];
    const float* ln1_b    = (const float*)d_in[2];
    const float* wqkv     = (const float*)d_in[3];
    const float* bqkv     = (const float*)d_in[4];
    const float* attnb    = (const float*)d_in[5];
    const int*   bidx     = (const int*)d_in[6];
    const float* wproj    = (const float*)d_in[7];
    const float* bproj    = (const float*)d_in[8];
    const float* conv_w   = (const float*)d_in[9];
    const float* bn_g     = (const float*)d_in[10];
    const float* bn_b     = (const float*)d_in[11];
    const float* bn_m     = (const float*)d_in[12];
    const float* bn_v     = (const float*)d_in[13];
    const float* ln2_g    = (const float*)d_in[14];
    const float* ln2_b    = (const float*)d_in[15];
    const float* w1       = (const float*)d_in[16];
    const float* b1       = (const float*)d_in[17];
    const float* w2       = (const float*)d_in[18];
    const float* b2       = (const float*)d_in[19];

    int n_off = in_sizes[5] / NHEAD;

    char* ws = (char*)d_ws;
    u16*   qkv = (u16*)(ws + OFF_QKV);
    u16*   xn  = (u16*)(ws + OFF_XN);
    u16*   o_  = xn;                 // overlay: xn dead after QKV GEMM
    float* x1  = (float*)(ws + OFF_X1);
    float* x2  = (float*)(ws + OFF_X2);
    u16*   mm  = (u16*)(ws + OFF_M);
    u16*   hh  = (u16*)(ws + 0);     // overlay: qkv + xn regions dead by MLP time
    float* out = (float*)d_out;

    // 1) LN1 + window partition -> xn (bf16, window order)
    ln_kernel<<<NTOK, 128, 0, stream>>>(x, ln1_g, ln1_b, xn, 1);
    // 2) QKV GEMM: 50176 x 1152 = xn @ wqkv + bqkv
    gemm_bf16<0><<<dim3(1152 / 128, NTOK / 128), 256, 0, stream>>>(
        xn, wqkv, bqkv, qkv, nullptr, NTOK, 1152, CDIM);
    // 3) windowed attention per (window, head)
    attn_kernel<<<dim3(NWIN, NHEAD), 128, 0, stream>>>(qkv, attnb, bidx, o_, n_off);
    // 4) proj GEMM + un-window + residual -> x1 (f32, token order)
    gemm_bf16<1><<<dim3(CDIM / 128, NTOK / 128), 256, 0, stream>>>(
        o_, wproj, bproj, x1, x, NTOK, CDIM, CDIM);
    // 5) depthwise conv + BN -> x2
    conv_bn_kernel<<<(NTOK * CDIM) / 256, 256, 0, stream>>>(x1, conv_w, bn_g, bn_b,
                                                            bn_m, bn_v, x2);
    // 6) LN2 -> mm (bf16)
    ln_kernel<<<NTOK, 128, 0, stream>>>(x2, ln2_g, ln2_b, mm, 0);
    // 7) MLP1 GEMM + exact GELU -> hh (bf16)
    gemm_bf16<2><<<dim3(HIDD / 128, NTOK / 128), 256, 0, stream>>>(
        mm, w1, b1, hh, nullptr, NTOK, HIDD, CDIM);
    // 8) MLP2 GEMM + residual -> out (f32)
    gemm_bf16<3><<<dim3(CDIM / 128, NTOK / 128), 256, 0, stream>>>(
        hh, w2, b2, out, x2, NTOK, CDIM, HIDD);
}
